// EPA_Layer_77807627534498
// MI455X (gfx1250) — matrix operand order repaired and checked by reading, not compiled, not run on hardware
//
#include <hip/hip_runtime.h>
#include <hip/hip_bf16.h>

// ---------------------------------------------------------------------------
// EPA layer for MI455X (gfx1250, wave32).  All contractions via
// v_wmma_f32_16x16x32_bf16 (bf16 operands, fp32 accumulate).  Channel-last
// layouts for softmax / SE / LayerNorm.  Pipeline is HBM-bandwidth bound, so
// WMMA-feeding intermediates are stored bf16 to halve traffic.  bf16 tile
// staging uses the CDNA5 direct global->LDS async path (ASYNCcnt) instead of
// a VGPR round trip; B-fragments are double-buffered around the WMMAs.
// ---------------------------------------------------------------------------

typedef __bf16 bf16;
typedef __attribute__((ext_vector_type(16))) __bf16 v16bf;
typedef __attribute__((ext_vector_type(8)))  float  v8f;

#define WMMA_BF16(acc, a, b)                                                   \
  acc = __builtin_amdgcn_wmma_f32_16x16x32_bf16(false, (a), false, (b),        \
                                                (short)0, (acc), false, false)

// K index of element e of the 16x32 bf16 A-fragment for a given lane
// (ISA 7.12.2: lanes 0-15 hold K 0-7,16-23; lanes 16-31 hold K 8-15,24-31)
__device__ __forceinline__ int a_k(int lane, int e) {
  int kb = (lane & 16) ? 8 : 0;
  int v = e >> 1, p = e & 1;
  return kb + ((v < 4) ? (2 * v + p) : (16 + 2 * (v - 4) + p));
}
// K index of element e of the 32x16 bf16 B-fragment (lane holds column n=lane&15;
// same K order as A: lanes 0-15: K 0-7,16-23; lanes 16-31: K 8-15,24-31)
__device__ __forceinline__ int b_k(int lane, int e) {
  return ((lane & 16) ? 8 : 0) + ((e < 8) ? e : (8 + e));
}

// Direct global -> LDS async copy, 16B per lane (GLOBAL_LOAD_ASYNC_TO_LDS_B128,
// ASYNCcnt-tracked; LDS byte address = low 32 bits of the generic pointer).
__device__ __forceinline__ void async_b128(void* lds_dst, const void* gsrc) {
  asm volatile("global_load_async_to_lds_b128 %0, %1, off"
               :
               : "v"((unsigned)(uintptr_t)lds_dst), "v"(gsrc)
               : "memory");
}
__device__ __forceinline__ void wait_async0() {
  asm volatile("s_wait_asynccnt 0x0" ::: "memory");
}

// ---------------------------------------------------------------------------
// weight prep
// ---------------------------------------------------------------------------
__global__ void cvt_bf16_kernel(const float* __restrict__ in,
                                bf16* __restrict__ out, int n) {
  int i = blockIdx.x * 256 + threadIdx.x;
  if (i < n) out[i] = (bf16)in[i];
}

// Wconv [oc][ic][3][3] fp32 -> [tap(kh*3+kw)][oc][ic] bf16
__global__ void pack_conv_kernel(const float* __restrict__ W,
                                 bf16* __restrict__ out) {
  int i = blockIdx.x * 256 + threadIdx.x;
  if (i >= 9 * 65536) return;
  int ic = i & 255, oc = (i >> 8) & 255, tap = i >> 16;
  int kh = tap / 3, kw = tap % 3;
  out[i] = (bf16)W[(((size_t)oc * 256 + ic) * 3 + kh) * 3 + kw];
}

// Wconvt [ic][oc][4][4] fp32 -> flipped/transposed Wt: [tap(kh*4+kw)][oc][ic]
__global__ void pack_convt_kernel(const float* __restrict__ W,
                                  bf16* __restrict__ out) {
  int i = blockIdx.x * 256 + threadIdx.x;
  if (i >= 16 * 65536) return;
  int ic = i & 255, oc = (i >> 8) & 255, tap = i >> 16;
  int kh = tap >> 2, kw = tap & 3;
  out[i] = (bf16)W[(((size_t)ic * 256 + oc) * 4 + (3 - kh)) * 4 + (3 - kw)];
}

// ---------------------------------------------------------------------------
// Projection GEMM: OUT[m, d] = sum_c X[m,c] * W[d,c]
// M = 65536 (pixels), N = K = 256.  Block: 256 thr (8 waves), M-tile 64,
// full N=256 per block.  Outputs either fp32 NHWC or bf16 NCHW.
// ---------------------------------------------------------------------------
__global__ __launch_bounds__(256) void gemm_proj_kernel(
    const float* __restrict__ X, const bf16* __restrict__ W,
    float* __restrict__ out_f, bf16* __restrict__ out_b, int nchw_bf16) {
  __shared__ bf16 Xl[64][32];
  __shared__ bf16 Wl[256][32];
  const int tid = threadIdx.x, lane = tid & 31, wv = tid >> 5;
  const int m0 = blockIdx.x * 64;
  const int mt = wv >> 1, nbase = (wv & 1) * 8;
  v8f acc[8] = {};

  for (int k0 = 0; k0 < 256; k0 += 32) {
    {  // stage W chunk 256x32 via async global->LDS (thread t -> row oc=t)
      const bf16* src = W + ((size_t)tid << 8) + k0;
      bf16* dst = &Wl[tid][0];
#pragma unroll
      for (int i = 0; i < 4; ++i) async_b128(dst + i * 8, src + i * 8);
    }
    {  // stage X tile 64x32 (8 elems / thread), fp32 -> bf16 via VALU
      int i0 = tid * 8, row = i0 >> 5, cc = i0 & 31;
      const float* src = X + (size_t)(m0 + row) * 256 + k0 + cc;
      if (k0 < 224) __builtin_prefetch(src + 32, 0, 0);
#pragma unroll
      for (int i = 0; i < 8; ++i) Xl[row][cc + i] = (bf16)src[i];
    }
    wait_async0();
    __syncthreads();
    v16bf a, bbuf[2];
    int m = mt * 16 + (lane & 15);
#pragma unroll
    for (int e = 0; e < 16; ++e) a[e] = Xl[m][a_k(lane, e)];
    {
      int n = nbase * 16 + (lane & 15);
#pragma unroll
      for (int e = 0; e < 16; ++e) bbuf[0][e] = Wl[n][b_k(lane, e)];
    }
#pragma unroll
    for (int j = 0; j < 8; ++j) {
      if (j < 7) {  // double-buffer next B fragment
        int n = (nbase + j + 1) * 16 + (lane & 15);
#pragma unroll
        for (int e = 0; e < 16; ++e) bbuf[(j + 1) & 1][e] = Wl[n][b_k(lane, e)];
      }
      WMMA_BF16(acc[j], a, bbuf[j & 1]);
    }
    __syncthreads();
  }

  int mrow0 = m0 + mt * 16 + ((lane & 16) ? 8 : 0);
#pragma unroll
  for (int j = 0; j < 8; ++j) {
    int n = (nbase + j) * 16 + (lane & 15);
#pragma unroll
    for (int r = 0; r < 8; ++r) {
      int m = mrow0 + r;
      if (nchw_bf16) {
        int b = m >> 12, pix = m & 4095;
        out_b[(((size_t)b * 256 + n) << 12) + pix] = (bf16)acc[j][r];
      } else {
        out_f[((size_t)m << 8) + n] = acc[j][r];
      }
    }
  }
}

// ---------------------------------------------------------------------------
// Attention logits per (b,c) slice.  Q,K are 64x64 bf16 in LDS.
// waves 0-7 : attn_s[h,g] = Q * K^T   -> lg_s[b,h,g,c]  (channel-last)
// waves 8-15: attn_c[w,v] = Q^T * K   -> lg_c[b,w,v,c]
// ---------------------------------------------------------------------------
__global__ __launch_bounds__(512) void attn_kernel(
    const bf16* __restrict__ q, const bf16* __restrict__ k,
    float* __restrict__ lg_s, float* __restrict__ lg_c) {
  __shared__ bf16 Ql[64][64];
  __shared__ bf16 Kl[64][64];
  const int tid = threadIdx.x, lane = tid & 31, wv = tid >> 5;
  const int bc = blockIdx.x;  // b*256 + c
  const int b = bc >> 8, c = bc & 255;
  const size_t base = (size_t)bc << 12;

  {  // async-stage both 64x64 slices (16B / thread / tensor)
    int i0 = tid * 8, r = i0 >> 6, cc = i0 & 63;
    async_b128(&Ql[r][cc], q + base + i0);
    async_b128(&Kl[r][cc], k + base + i0);
  }
  wait_async0();
  __syncthreads();

  if (wv < 8) {  // spatial attention: A=Q rows(h), B=K rows(g), K-dim = w
#pragma unroll
    for (int t2 = 0; t2 < 2; ++t2) {
      int t = wv + t2 * 8;
      int ht = t >> 2, gt = t & 3;
      v8f acc = {};
#pragma unroll
      for (int kk = 0; kk < 2; ++kk) {
        int k0 = kk * 32;
        v16bf a, bb;
        int m = ht * 16 + (lane & 15);
#pragma unroll
        for (int e = 0; e < 16; ++e) a[e] = Ql[m][k0 + a_k(lane, e)];
        int n = gt * 16 + (lane & 15);
#pragma unroll
        for (int e = 0; e < 16; ++e) bb[e] = Kl[n][k0 + b_k(lane, e)];
        WMMA_BF16(acc, a, bb);
      }
      int g = gt * 16 + (lane & 15);
      int h0 = ht * 16 + ((lane & 16) ? 8 : 0);
#pragma unroll
      for (int r = 0; r < 8; ++r)
        lg_s[(((size_t)(b * 64 + h0 + r) * 64 + g) << 8) + c] = acc[r];
    }
  } else {  // channel attention: A=Q^T (m=w, K-dim=h), B=K cols(v)
    int w2 = wv - 8;
#pragma unroll
    for (int t2 = 0; t2 < 2; ++t2) {
      int t = w2 + t2 * 8;
      int wt = t >> 2, vt = t & 3;
      v8f acc = {};
#pragma unroll
      for (int kk = 0; kk < 2; ++kk) {
        int k0 = kk * 32;
        v16bf a, bb;
        int m = wt * 16 + (lane & 15);
#pragma unroll
        for (int e = 0; e < 16; ++e) a[e] = Ql[k0 + a_k(lane, e)][m];
        int n = vt * 16 + (lane & 15);
#pragma unroll
        for (int e = 0; e < 16; ++e) bb[e] = Kl[k0 + b_k(lane, e)][n];
        WMMA_BF16(acc, a, bb);
      }
      int v = vt * 16 + (lane & 15);
      int w0 = wt * 16 + ((lane & 16) ? 8 : 0);
#pragma unroll
      for (int r = 0; r < 8; ++r)
        lg_c[(((size_t)(b * 64 + w0 + r) * 64 + v) << 8) + c] = acc[r];
    }
  }
}

// ---------------------------------------------------------------------------
// Channel softmax: one wave per contiguous 256-float row.
// mode: 0 -> fp32 out, 1 -> bf16 out.
// ---------------------------------------------------------------------------
__global__ __launch_bounds__(256) void softmax_kernel(
    const float* __restrict__ in, float* __restrict__ of,
    bf16* __restrict__ ob, int to_bf16) {
  const int lane = threadIdx.x & 31, wv = threadIdx.x >> 5;
  const size_t row = (size_t)blockIdx.x * 8 + wv;
  const float* p = in + (row << 8);
  float v[8];
#pragma unroll
  for (int i = 0; i < 8; ++i) v[i] = p[lane + i * 32];
  float m = v[0];
#pragma unroll
  for (int i = 1; i < 8; ++i) m = fmaxf(m, v[i]);
#pragma unroll
  for (int s = 16; s; s >>= 1) m = fmaxf(m, __shfl_xor(m, s, 32));
  float sum = 0.f;
#pragma unroll
  for (int i = 0; i < 8; ++i) {
    v[i] = __expf(v[i] - m);
    sum += v[i];
  }
#pragma unroll
  for (int s = 16; s; s >>= 1) sum += __shfl_xor(sum, s, 32);
  float inv = 1.0f / sum;
  if (to_bf16) {
    bf16* o = ob + (row << 8);
#pragma unroll
    for (int i = 0; i < 8; ++i) o[lane + i * 32] = (bf16)(v[i] * inv);
  } else {
    float* o = of + (row << 8);
#pragma unroll
    for (int i = 0; i < 8; ++i) o[lane + i * 32] = v[i] * inv;
  }
}

// ---------------------------------------------------------------------------
// conv3x3 stride2 pad1 + ReLU as 9 tap-GEMMs.
// In : attn_s softmax, bf16 [B,64,64,256] channel-last.
// Out: bf16 [B,32,32,256].
// One block per (b, oh): M = 32 ow positions, N = 256 oc, K = 256 per tap.
// ---------------------------------------------------------------------------
__global__ __launch_bounds__(256) void conv3x3_kernel(
    const bf16* __restrict__ A, const bf16* __restrict__ Wc,
    bf16* __restrict__ R) {
  __shared__ bf16 Al[32][32];
  __shared__ bf16 Wl[256][32];
  const int tid = threadIdx.x, lane = tid & 31, wv = tid >> 5;
  const int b = blockIdx.x >> 5, oh = blockIdx.x & 31;
  const int nt0 = wv * 2;
  v8f acc[2][2] = {};

  for (int kh = 0; kh < 3; ++kh) {
    int ih = 2 * oh - 1 + kh;
    bool vh = (ih >= 0) && (ih < 64);
    for (int kw = 0; kw < 3; ++kw) {
      for (int kc = 0; kc < 8; ++kc) {
        int k0 = kc * 32;
        {  // async-stage W tap chunk: 256 oc x 32 ic (thread t -> oc=t)
          const bf16* src =
              Wc + (((size_t)(kh * 3 + kw) * 256 + tid) << 8) + k0;
          bf16* dst = &Wl[tid][0];
#pragma unroll
          for (int i = 0; i < 4; ++i) async_b128(dst + i * 8, src + i * 8);
        }
        {  // stage A: 32 ow x 32 ic (4 / thread), boundary zero-fill
          int i0 = tid * 4, ow = i0 >> 5, ic = i0 & 31;
          int iw = 2 * ow - 1 + kw;
          bool ok = vh && (iw >= 0) && (iw < 64);
          const bf16* src =
              A + ((((size_t)b * 64 + (ok ? ih : 0)) * 64 + (ok ? iw : 0)) << 8) +
              k0 + ic;
          if (ok && kc < 7) __builtin_prefetch(src + 32, 0, 0);
#pragma unroll
          for (int i = 0; i < 4; ++i) Al[ow][ic + i] = ok ? src[i] : (bf16)0.0f;
        }
        wait_async0();
        __syncthreads();
        v16bf a0, a1, bb0, bb1;
        int mloc = lane & 15;
#pragma unroll
        for (int e = 0; e < 16; ++e) a0[e] = Al[mloc][a_k(lane, e)];
#pragma unroll
        for (int e = 0; e < 16; ++e) a1[e] = Al[16 + mloc][a_k(lane, e)];
        {
          int n0 = nt0 * 16 + (lane & 15), n1 = (nt0 + 1) * 16 + (lane & 15);
#pragma unroll
          for (int e = 0; e < 16; ++e) bb0[e] = Wl[n0][b_k(lane, e)];
#pragma unroll
          for (int e = 0; e < 16; ++e) bb1[e] = Wl[n1][b_k(lane, e)];
        }
        WMMA_BF16(acc[0][0], a0, bb0);
        WMMA_BF16(acc[1][0], a1, bb0);
        WMMA_BF16(acc[0][1], a0, bb1);
        WMMA_BF16(acc[1][1], a1, bb1);
        __syncthreads();
      }
    }
  }
#pragma unroll
  for (int mt = 0; mt < 2; ++mt) {
    int ow0 = mt * 16 + ((lane & 16) ? 8 : 0);
#pragma unroll
    for (int j = 0; j < 2; ++j) {
      int oc = (nt0 + j) * 16 + (lane & 15);
#pragma unroll
      for (int r = 0; r < 8; ++r) {
        float v = acc[mt][j][r];
        R[((((size_t)b * 32 + oh) * 32 + (ow0 + r)) << 8) + oc] =
            (bf16)fmaxf(v, 0.0f);
      }
    }
  }
}

// ---------------------------------------------------------------------------
// convT 4x4 stride2 pad1 + ReLU as 4 parity-split tap-GEMMs per output row.
// In : R bf16 [B,32,32,256].  Out: S fp32 [B,64,64,256] channel-last.
// One block per (b, y).  M = 64 x positions split into even/odd halves.
// ---------------------------------------------------------------------------
__global__ __launch_bounds__(256) void convt4x4_kernel(
    const bf16* __restrict__ R, const bf16* __restrict__ Wt,
    float* __restrict__ S) {
  __shared__ bf16 Al[2][32][32];   // [x-parity][pos][ic]
  __shared__ bf16 Wl[2][256][32];  // [x-parity][oc][ic]
  const int tid = threadIdx.x, lane = tid & 31, wv = tid >> 5;
  const int b = blockIdx.x >> 6, y = blockIdx.x & 63;
  const int mt = wv >> 1;    // 0..3: tiles 0-1 even-x, 2-3 odd-x
  const int hx = mt >> 1;    // x parity handled by this wave
  const int nbase = (wv & 1) * 8;
  v8f acc[8] = {};

  for (int khi = 0; khi < 2; ++khi) {
    int kh = 2 * khi + (y & 1);
    int ih = ((y + 1) >> 1) + khi - 1;
    if (ih < 0 || ih >= 32) continue;  // uniform across block
    for (int kwi = 0; kwi < 2; ++kwi) {
      for (int kc = 0; kc < 8; ++kc) {
        int k0 = kc * 32;
        {  // async-stage W both parities: 2*256*32 (128B / thread)
          int hh = tid >> 7;
          int oc0 = (tid & 127) * 2;
          int kw = 2 * kwi + hh;
#pragma unroll
          for (int rr = 0; rr < 2; ++rr) {
            const bf16* src =
                Wt + (((size_t)(kh * 4 + kw) * 256 + oc0 + rr) << 8) + k0;
            bf16* dst = &Wl[hh][oc0 + rr][0];
#pragma unroll
            for (int i = 0; i < 4; ++i) async_b128(dst + i * 8, src + i * 8);
          }
        }
        {  // stage A both parities: 2*32*32 (8 / thread), zero-fill edges
          int i0 = tid * 8;
          int hh = i0 >> 10, j = (i0 >> 5) & 31, ic = i0 & 31;
          int d = (hh == 0) ? (kwi - 1) : kwi;
          int iw = j + d;
          bool ok = (iw >= 0) && (iw < 32);
          const bf16* src =
              R + ((((size_t)b * 32 + ih) * 32 + (ok ? iw : 0)) << 8) + k0 + ic;
#pragma unroll
          for (int i = 0; i < 8; ++i)
            Al[hh][j][ic + i] = ok ? src[i] : (bf16)0.0f;
        }
        wait_async0();
        __syncthreads();
        v16bf a, bbuf[2];
        int m = (mt & 1) * 16 + (lane & 15);
#pragma unroll
        for (int e = 0; e < 16; ++e) a[e] = Al[hx][m][a_k(lane, e)];
        {
          int n = nbase * 16 + (lane & 15);
#pragma unroll
          for (int e = 0; e < 16; ++e) bbuf[0][e] = Wl[hx][n][b_k(lane, e)];
        }
#pragma unroll
        for (int j = 0; j < 8; ++j) {
          if (j < 7) {  // double-buffer next B fragment
            int n = (nbase + j + 1) * 16 + (lane & 15);
#pragma unroll
            for (int e = 0; e < 16; ++e)
              bbuf[(j + 1) & 1][e] = Wl[hx][n][b_k(lane, e)];
          }
          WMMA_BF16(acc[j], a, bbuf[j & 1]);
        }
        __syncthreads();
      }
    }
  }
  int row0 = (mt & 1) * 16 + ((lane & 16) ? 8 : 0);
#pragma unroll
  for (int j = 0; j < 8; ++j) {
    int oc = (nbase + j) * 16 + (lane & 15);
#pragma unroll
    for (int r = 0; r < 8; ++r) {
      int x = 2 * (row0 + r) + hx;
      S[((((size_t)b * 64 + y) * 64 + x) << 8) + oc] = fmaxf(acc[j][r], 0.0f);
    }
  }
}

// ---------------------------------------------------------------------------
// SE gate: g = mean over (w,v) of attn_c; gate = sigmoid(relu(g Wc1^T) Wc2^T).
// One block per batch.
// ---------------------------------------------------------------------------
__global__ __launch_bounds__(256) void se_kernel(
    const float* __restrict__ AC, const float* __restrict__ Wc1,
    const float* __restrict__ Wc2, float* __restrict__ gate) {
  __shared__ float g[256];
  __shared__ float h[64];
  const int b = blockIdx.x, t = threadIdx.x;
  float acc = 0.f;
  const float* p = AC + ((size_t)b << 20) + t;
  for (int pos = 0; pos < 4096; ++pos) acc += p[(size_t)pos << 8];
  g[t] = acc * (1.0f / 4096.0f);
  __syncthreads();
  if (t < 64) {
    float s = 0.f;
    const float* w = Wc1 + t * 256;
    for (int c = 0; c < 256; ++c) s += w[c] * g[c];
    h[t] = fmaxf(s, 0.0f);
  }
  __syncthreads();
  float s = 0.f;
  const float* w = Wc2 + t * 64;
  for (int j = 0; j < 64; ++j) s += w[j] * h[j];
  gate[b * 256 + t] = 1.0f / (1.0f + __expf(-s));
}

// ---------------------------------------------------------------------------
// Fused residual + gating + LayerNorm.  One wave per pixel (256 channels).
// ---------------------------------------------------------------------------
__global__ __launch_bounds__(256) void final_ln_kernel(
    const float* __restrict__ x, const float* __restrict__ vs,
    const float* __restrict__ vc, const float* __restrict__ S,
    const float* __restrict__ AC, const float* __restrict__ gate,
    const float* __restrict__ gamma, const float* __restrict__ beta,
    float* __restrict__ out) {
  const int lane = threadIdx.x & 31, wv = threadIdx.x >> 5;
  const size_t p = (size_t)blockIdx.x * 8 + wv;
  const int b = (int)(p >> 12);
  const size_t base = p << 8;
  float y[8];
#pragma unroll
  for (int i = 0; i < 8; ++i) {
    int c = lane + i * 32;
    float s1 = S[base + c] * vs[base + c];
    float c1 = gate[b * 256 + c] * AC[base + c] * vc[base + c];
    y[i] = x[base + c] + s1 + c1;
  }
  float sum = 0.f;
#pragma unroll
  for (int i = 0; i < 8; ++i) sum += y[i];
#pragma unroll
  for (int s = 16; s; s >>= 1) sum += __shfl_xor(sum, s, 32);
  float mu = sum * (1.0f / 256.0f);
  float vsum = 0.f;
#pragma unroll
  for (int i = 0; i < 8; ++i) {
    float d = y[i] - mu;
    vsum += d * d;
  }
#pragma unroll
  for (int s = 16; s; s >>= 1) vsum += __shfl_xor(vsum, s, 32);
  float rs = rsqrtf(vsum * (1.0f / 256.0f) + 1e-5f);
#pragma unroll
  for (int i = 0; i < 8; ++i) {
    int c = lane + i * 32;
    out[base + c] = (y[i] - mu) * rs * gamma[c] + beta[c];
  }
}

// ---------------------------------------------------------------------------
// launcher
// ---------------------------------------------------------------------------
extern "C" void kernel_launch(void* const* d_in, const int* in_sizes, int n_in,
                              void* d_out, int out_size, void* d_ws,
                              size_t ws_size, hipStream_t stream) {
  (void)in_sizes; (void)n_in; (void)out_size; (void)ws_size;
  const float* x      = (const float*)d_in[0];
  const float* Wq     = (const float*)d_in[1];
  const float* Wk     = (const float*)d_in[2];
  const float* Wvc    = (const float*)d_in[3];
  const float* Wvs    = (const float*)d_in[4];
  const float* Wc1    = (const float*)d_in[5];
  const float* Wc2    = (const float*)d_in[6];
  const float* Wconv  = (const float*)d_in[7];
  const float* Wconvt = (const float*)d_in[8];
  const float* gamma  = (const float*)d_in[9];
  const float* beta   = (const float*)d_in[10];
  float* out = (float*)d_out;

  char* ws = (char*)d_ws;
  size_t off = 0;
  auto alloc = [&](size_t bytes) {
    size_t r = off;
    off += (bytes + 255) & ~(size_t)255;
    return r;
  };
  const size_t SZ_BF = (size_t)16 * 256 * 4096 * 2;  // 32 MiB  (bf16 tensor)
  const size_t SZ_F  = (size_t)16 * 256 * 4096 * 4;  // 64 MiB  (fp32 tensor)

  bf16*  q_b   = (bf16*)(ws + alloc(SZ_BF));
  bf16*  k_b   = (bf16*)(ws + alloc(SZ_BF));
  float* vc    = (float*)(ws + alloc(SZ_F));
  float* vs    = (float*)(ws + alloc(SZ_F));
  float* lg_s  = (float*)(ws + alloc(SZ_F));  // reused as S after conv chain
  float* lg_c  = (float*)(ws + alloc(SZ_F));
  bf16*  as_b  = (bf16*)(ws + alloc(SZ_BF));
  float* ac_f  = (float*)(ws + alloc(SZ_F));
  bf16*  r_b   = (bf16*)(ws + alloc((size_t)16 * 1024 * 256 * 2));
  bf16*  wq_b  = (bf16*)(ws + alloc(65536 * 2));
  bf16*  wk_b  = (bf16*)(ws + alloc(65536 * 2));
  bf16*  wvc_b = (bf16*)(ws + alloc(65536 * 2));
  bf16*  wvs_b = (bf16*)(ws + alloc(65536 * 2));
  bf16*  wc_b  = (bf16*)(ws + alloc((size_t)9 * 65536 * 2));
  bf16*  wt_b  = (bf16*)(ws + alloc((size_t)16 * 65536 * 2));
  float* gate  = (float*)(ws + alloc(16 * 256 * 4));
  float* S     = lg_s;  // convT output aliases dead logits_s buffer

  // 1. weight prep
  cvt_bf16_kernel<<<256, 256, 0, stream>>>(Wq, wq_b, 65536);
  cvt_bf16_kernel<<<256, 256, 0, stream>>>(Wk, wk_b, 65536);
  cvt_bf16_kernel<<<256, 256, 0, stream>>>(Wvc, wvc_b, 65536);
  cvt_bf16_kernel<<<256, 256, 0, stream>>>(Wvs, wvs_b, 65536);
  pack_conv_kernel<<<(9 * 65536) / 256, 256, 0, stream>>>(Wconv, wc_b);
  pack_convt_kernel<<<(16 * 65536) / 256, 256, 0, stream>>>(Wconvt, wt_b);

  // 2. projections (WMMA)
  gemm_proj_kernel<<<1024, 256, 0, stream>>>(x, wq_b, nullptr, q_b, 1);
  gemm_proj_kernel<<<1024, 256, 0, stream>>>(x, wk_b, nullptr, k_b, 1);
  gemm_proj_kernel<<<1024, 256, 0, stream>>>(x, wvc_b, vc, nullptr, 0);
  gemm_proj_kernel<<<1024, 256, 0, stream>>>(x, wvs_b, vs, nullptr, 0);

  // 3. attention logits (WMMA), channel-last
  attn_kernel<<<4096, 512, 0, stream>>>(q_b, k_b, lg_s, lg_c);

  // 4. channel softmax
  softmax_kernel<<<8192, 256, 0, stream>>>(lg_s, nullptr, as_b, 1);
  softmax_kernel<<<8192, 256, 0, stream>>>(lg_c, ac_f, nullptr, 0);

  // 5. conv3x3 s2 + ReLU (WMMA)
  conv3x3_kernel<<<16 * 32, 256, 0, stream>>>(as_b, wc_b, r_b);

  // 6. convT4x4 s2 + ReLU (WMMA)
  convt4x4_kernel<<<16 * 64, 256, 0, stream>>>(r_b, wt_b, S);

  // 7. SE gate
  se_kernel<<<16, 256, 0, stream>>>(ac_f, Wc1, Wc2, gate);

  // 8. fused residual + gating + LayerNorm
  final_ln_kernel<<<8192, 256, 0, stream>>>(x, vs, vc, S, ac_f, gate, gamma,
                                            beta, out);
}
